// KDEAttention_69801808494919
// MI455X (gfx1250) — matrix operand order repaired and checked by reading, not compiled, not run on hardware
//
#include <hip/hip_runtime.h>
#include <hip/hip_bf16.h>

typedef __attribute__((ext_vector_type(16))) _Float16 v16h;
typedef __attribute__((ext_vector_type(8)))  float    v8f;

#define CH_STRIDE 16384   // 128*128 spatial elements per channel
#define ROW_STRIDE 128

// ---------------------------------------------------------------------------
// Fragment loaders for V_WMMA_F32_16X16X32_F16 (wave32), built straight from
// global memory (fp32 -> f16 convert on the fly).
//
// A (16x32, M x K) layout per ISA 7.12.2:
//   lanes 0-15 : M = lane,    halves 0..7 = K 0..7,  halves 8..15 = K 16..23
//   lanes 16-31: M = lane-16, halves 0..7 = K 8..15, halves 8..15 = K 24..31
// With L = i*16 + j (i = spatial row in tile, j = col), K base = 32*s:
//   halves 0..7  -> row 2s,   cols (lane>>4)*8 + 0..7
//   halves 8..15 -> row 2s+1, cols (lane>>4)*8 + 0..7
// ---------------------------------------------------------------------------
__device__ inline v16h load_a_frag(const float* __restrict__ base, int i0,
                                   int s, int lane) {
  const int m    = i0 + (lane & 15);
  const int half = lane >> 4;
  const float* p = base + (size_t)m * CH_STRIDE + (2 * s) * ROW_STRIDE + half * 8;
  float4 f0 = *(const float4*)(p);
  float4 f1 = *(const float4*)(p + 4);
  float4 f2 = *(const float4*)(p + ROW_STRIDE);
  float4 f3 = *(const float4*)(p + ROW_STRIDE + 4);
  v16h a;
  a[0]  = (_Float16)f0.x; a[1]  = (_Float16)f0.y;
  a[2]  = (_Float16)f0.z; a[3]  = (_Float16)f0.w;
  a[4]  = (_Float16)f1.x; a[5]  = (_Float16)f1.y;
  a[6]  = (_Float16)f1.z; a[7]  = (_Float16)f1.w;
  a[8]  = (_Float16)f2.x; a[9]  = (_Float16)f2.y;
  a[10] = (_Float16)f2.z; a[11] = (_Float16)f2.w;
  a[12] = (_Float16)f3.x; a[13] = (_Float16)f3.y;
  a[14] = (_Float16)f3.z; a[15] = (_Float16)f3.w;
  return a;
}

// B (32x16, K x N) = transpose of the J-block rows of A (gram = A*A^T).
// Layout: lanes n and 16+n hold column n of B in AMD's table K order;
// same K order as A. So lane (hb*16+n) needs
// channel row (j0+n), halves 0..7 -> spatial row 2s, cols hb*8 + 0..7;
// halves 8..15 -> spatial row 2s+1, cols hb*8 + 0..7.
__device__ inline v16h load_b_frag(const float* __restrict__ base, int j0,
                                   int s, int lane) {
  const int n  = lane & 15;
  const int hb = lane >> 4;
  const float* p = base + (size_t)(j0 + n) * CH_STRIDE + (2 * s) * ROW_STRIDE + hb * 8;
  float4 f0 = *(const float4*)(p);
  float4 f1 = *(const float4*)(p + 4);
  float4 f2 = *(const float4*)(p + ROW_STRIDE);
  float4 f3 = *(const float4*)(p + ROW_STRIDE + 4);
  v16h b;
  b[0]  = (_Float16)f0.x; b[1]  = (_Float16)f0.y;
  b[2]  = (_Float16)f0.z; b[3]  = (_Float16)f0.w;
  b[4]  = (_Float16)f1.x; b[5]  = (_Float16)f1.y;
  b[6]  = (_Float16)f1.z; b[7]  = (_Float16)f1.w;
  b[8]  = (_Float16)f2.x; b[9]  = (_Float16)f2.y;
  b[10] = (_Float16)f2.z; b[11] = (_Float16)f2.w;
  b[12] = (_Float16)f3.x; b[13] = (_Float16)f3.y;
  b[14] = (_Float16)f3.z; b[15] = (_Float16)f3.w;
  return b;
}

__device__ inline v8f wmma_f16(v16h a, v16h b, v8f c) {
  // (neg_a, A, neg_b, B, c_mod, C, reuse_a, reuse_b)
  return __builtin_amdgcn_wmma_f32_16x16x32_f16(false, a, false, b,
                                                (short)0, c, false, false);
}

// ---------------------------------------------------------------------------
// Kernel 1: per-tile Gram (32x32 = 2x2 blocks of 16x16, 8 k-steps of K=32)
// + KDE density. One wave per tile; 8 waves per block; 1024 tiles total.
// dist2 derived entirely from the Gram so the diagonal is exactly 0.
// ---------------------------------------------------------------------------
__global__ __launch_bounds__(256) void kde_gram_density_kernel(
    const float* __restrict__ x, float* __restrict__ dens) {
  __shared__ float lds_gram[8 * 32 * 32];  // 4KB per wave
  const int wave = threadIdx.x >> 5;
  const int lane = threadIdx.x & 31;
  const int tile = blockIdx.x * 8 + wave;  // [0, 1024)
  // tile -> (bi, g, w1, w2) per reshape (b, G, Wd, Wd)
  const int w2 = tile & 7;
  const int w1 = (tile >> 3) & 7;
  const int g  = (tile >> 6) & 3;
  const int bi = tile >> 8;

  const float* base = x + (size_t)bi * (128 * CH_STRIDE)
                        + (size_t)(g * 32) * CH_STRIDE
                        + (size_t)(w1 * 16) * ROW_STRIDE + w2 * 16;

  v8f c00 = {}, c01 = {}, c10 = {}, c11 = {};
#pragma unroll
  for (int s = 0; s < 8; ++s) {
    v16h a0 = load_a_frag(base, 0, s, lane);
    v16h a1 = load_a_frag(base, 16, s, lane);
    v16h b0 = load_b_frag(base, 0, s, lane);
    v16h b1 = load_b_frag(base, 16, s, lane);
    c00 = wmma_f16(a0, b0, c00);
    c01 = wmma_f16(a0, b1, c01);
    c10 = wmma_f16(a1, b0, c10);
    c11 = wmma_f16(a1, b1, c11);
  }

  // Spill Gram to this wave's LDS region. C/D layout: VGPR r -> M = r
  // (lanes 0-15) or M = 8+r (lanes 16-31); N = lane & 15.
  float* my = lds_gram + wave * (32 * 32);
  const int Mb = (lane < 16) ? 0 : 8;
  const int N  = lane & 15;
#pragma unroll
  for (int r = 0; r < 8; ++r) {
    my[(Mb + r) * 32 + N]            = c00[r];
    my[(Mb + r) * 32 + 16 + N]       = c01[r];
    my[(16 + Mb + r) * 32 + N]       = c10[r];
    my[(16 + Mb + r) * 32 + 16 + N]  = c11[r];
  }
  __syncthreads();

  // Lane ci computes density[ci] = mean_j exp(-0.5*(g_ii + g_jj - 2 g_ij))
  const int ci = lane;
  const float gii = my[ci * 32 + ci];
  float sum = 0.0f;
#pragma unroll
  for (int cj = 0; cj < 32; ++cj) {
    float d2 = gii + my[cj * 32 + cj] - 2.0f * my[ci * 32 + cj];
    d2 = fmaxf(d2, 0.0f);
    sum += __expf(-0.5f * d2);
  }
  // density layout: (b, C=G*c, W1, W2)
  dens[(((size_t)bi * 128 + g * 32 + ci) * 8 + w1) * 8 + w2] = sum * (1.0f / 32.0f);
}

// ---------------------------------------------------------------------------
// Kernel 2: 1x1 conv -> relu -> 1x1 conv -> sigmoid + GAMMA at 8x8 resolution.
// One block per (bi, hp, wp) position; 256 blocks of 128 threads.
// ---------------------------------------------------------------------------
__global__ __launch_bounds__(128) void kde_mlp_kernel(
    const float* __restrict__ dens,
    const float* __restrict__ w1, const float* __restrict__ b1,
    const float* __restrict__ w2, const float* __restrict__ b2,
    float* __restrict__ attn) {
  __shared__ float sd[128];
  __shared__ float sh[32];
  const int t  = threadIdx.x;
  const int wp = blockIdx.x & 7;
  const int hp = (blockIdx.x >> 3) & 7;
  const int bi = blockIdx.x >> 6;

  sd[t] = dens[(((size_t)bi * 128 + t) * 8 + hp) * 8 + wp];
  __syncthreads();

  if (t < 32) {
    float h = b1[t];
    const float* wr = w1 + t * 128;
#pragma unroll 8
    for (int c = 0; c < 128; ++c) h += wr[c] * sd[c];
    sh[t] = fmaxf(h, 0.0f);
  }
  __syncthreads();

  float y = b2[t];
  const float* wr = w2 + t * 32;
#pragma unroll
  for (int j = 0; j < 32; ++j) y += wr[j] * sh[j];
  const float a = 1.0f / (1.0f + __expf(-y)) + 0.5f;  // sigmoid + GAMMA
  attn[(((size_t)bi * 128 + t) * 8 + hp) * 8 + wp] = a;
}

// ---------------------------------------------------------------------------
// Kernel 3: half-pixel bilinear upsample (8x8 -> 128x128, edge clamp) of attn
// and elementwise multiply with x. One thread per output element.
// ---------------------------------------------------------------------------
__global__ __launch_bounds__(256) void kde_scale_kernel(
    const float* __restrict__ x, const float* __restrict__ attn,
    float* __restrict__ out) {
  const size_t idx = (size_t)blockIdx.x * blockDim.x + threadIdx.x;
  const int ww = idx & 127;
  const int hh = (idx >> 7) & 127;
  const int ch = (idx >> 14) & 127;
  const int bi = (int)(idx >> 21);

  // src = (dst + 0.5)/16 - 0.5 (scale factor 8/128)
  const float shf = (hh + 0.5f) * 0.0625f - 0.5f;
  const float swf = (ww + 0.5f) * 0.0625f - 0.5f;
  const float hf0 = floorf(shf);
  const float wf0 = floorf(swf);
  const float fh = shf - hf0;
  const float fw = swf - wf0;
  int h0 = (int)hf0, w0 = (int)wf0;
  int h1 = min(max(h0 + 1, 0), 7);
  int w1 = min(max(w0 + 1, 0), 7);
  h0 = min(max(h0, 0), 7);
  w0 = min(max(w0, 0), 7);

  const float* ap = attn + ((size_t)bi * 128 + ch) * 64;
  const float a00 = ap[h0 * 8 + w0];
  const float a01 = ap[h0 * 8 + w1];
  const float a10 = ap[h1 * 8 + w0];
  const float a11 = ap[h1 * 8 + w1];
  const float top = a00 + (a01 - a00) * fw;
  const float bot = a10 + (a11 - a10) * fw;
  const float a = top + (bot - top) * fh;

  out[idx] = x[idx] * a;
}

extern "C" void kernel_launch(void* const* d_in, const int* in_sizes, int n_in,
                              void* d_out, int out_size, void* d_ws, size_t ws_size,
                              hipStream_t stream) {
  const float* x  = (const float*)d_in[0];   // (4,128,128,128)
  const float* w1 = (const float*)d_in[1];   // (32,128)
  const float* b1 = (const float*)d_in[2];   // (32,)
  const float* w2 = (const float*)d_in[3];   // (128,32)
  const float* b2 = (const float*)d_in[4];   // (128,)
  float* out = (float*)d_out;

  float* dens = (float*)d_ws;                // 4*128*8*8 = 32768 floats
  float* attn = dens + 32768;                // 32768 floats

  // 1024 tiles, 8 waves (one tile each) per block
  kde_gram_density_kernel<<<128, 256, 0, stream>>>(x, dens);
  // 256 positions (b * 8 * 8)
  kde_mlp_kernel<<<256, 128, 0, stream>>>(dens, w1, b1, w2, b2, attn);
  // 8.4M output elements
  kde_scale_kernel<<<(4 * 128 * 128 * 128) / 256, 256, 0, stream>>>(x, attn, out);
}
